// Head_46952582479889
// MI455X (gfx1250) — matrix operand order repaired and checked by reading, not compiled, not run on hardware
//
#include <hip/hip_runtime.h>
#include <hip/hip_bf16.h>

typedef __attribute__((ext_vector_type(16))) __bf16 v16bf;
typedef __attribute__((ext_vector_type(8)))  __bf16 v8bf;
typedef __attribute__((ext_vector_type(8)))  float  v8f;
typedef __attribute__((ext_vector_type(4)))  int    v4i;

#define EMBED 1024
#define SEQ   2048
#define HD    64
#define NB    8

__device__ __forceinline__ v8bf ldg8(const __bf16* p) {
    return *reinterpret_cast<const v8bf*>(p);
}

__device__ __forceinline__ v16bf cat16(v8bf lo, v8bf hi) {
    v16bf r;
#pragma unroll
    for (int i = 0; i < 8; ++i) { r[i] = lo[i]; r[8 + i] = hi[i]; }
    return r;
}

__device__ __forceinline__ v8f wmma_bf16(v16bf a, v16bf b, v8f c) {
    return __builtin_amdgcn_wmma_f32_16x16x32_bf16(
        false, a, false, b, (short)0, c, false, false);
}

// gfx1250 async global->LDS copy (16B per lane), ASYNCcnt-tracked.
typedef v4i __attribute__((address_space(1)))* gptr_v4i;
typedef v4i __attribute__((address_space(3)))* lptr_v4i;

__device__ __forceinline__ void async_copy16(const __bf16* g, __bf16* l) {
#if __has_builtin(__builtin_amdgcn_global_load_async_to_lds_b128)
    __builtin_amdgcn_global_load_async_to_lds_b128(
        (gptr_v4i)(unsigned long long)(size_t)g,
        (lptr_v4i)(unsigned)(size_t)l,
        0, 0);
#else
    unsigned lds_off = (unsigned)(size_t)l;
    asm volatile("global_load_async_to_lds_b128 %0, %1, off"
                 :: "v"(lds_off), "v"(g) : "memory");
#endif
}

__device__ __forceinline__ void wait_async0() {
#if __has_builtin(__builtin_amdgcn_s_wait_asynccnt)
    __builtin_amdgcn_s_wait_asynccnt(0);
#else
    asm volatile("s_wait_asynccnt 0x0" ::: "memory");
#endif
}

__device__ __forceinline__ void wait_async4() {
#if __has_builtin(__builtin_amdgcn_s_wait_asynccnt)
    __builtin_amdgcn_s_wait_asynccnt(4);
#else
    asm volatile("s_wait_asynccnt 0x4" ::: "memory");
#endif
}

// ---------------------------------------------------------------------------
// Kernel 0: transpose W (E x D fp32) -> WT (D x E bf16), 3 matrices packed.
// ---------------------------------------------------------------------------
__global__ void wtrans_kernel(const float* __restrict__ Wq,
                              const float* __restrict__ Wk,
                              const float* __restrict__ Wv,
                              __bf16* __restrict__ WT) {
    int t = blockIdx.x * blockDim.x + threadIdx.x;   // 0 .. 3*64*1024-1
    int mat = t >> 16;
    int rem = t & 65535;
    int d = rem >> 10;        // 0..63
    int e = rem & 1023;       // 0..1023
    const float* W = (mat == 0) ? Wq : (mat == 1) ? Wk : Wv;
    WT[(size_t)t] = (__bf16)W[(size_t)e * HD + d];
}

// ---------------------------------------------------------------------------
// Kernel 1: QKV projection. One wave -> 16 rows x 64 cols of Q, K, V.
// All 12 B-fragments get private registers so loads clause up and the
// 12 WMMAs issue back-to-back.
// ---------------------------------------------------------------------------
__global__ __launch_bounds__(128)
void qkv_project_kernel(const float* __restrict__ X,
                        const __bf16* __restrict__ WT,   // [3][64][1024]
                        __bf16* __restrict__ Qb,         // [B][S][64]
                        __bf16* __restrict__ Kb,         // [B][S][64]
                        __bf16* __restrict__ Vt) {       // [B][64][S]
    int lane = threadIdx.x & 31;
    int wid  = threadIdx.x >> 5;
    int n    = lane & 15;
    int half = lane >> 4;

    int tile = blockIdx.x * 4 + wid;      // 0 .. B*S/16-1 = 1023
    int b  = tile >> 7;                   // /128
    int st = tile & 127;
    int s0 = st * 16;

    v8f acc[3][4] = {};

    const float* xrow = X + ((size_t)(b * SEQ + s0 + n)) * EMBED;  // row m = n
    const __bf16* wbase = WT + (size_t)n * EMBED + 16 * half;

    for (int kb = 0; kb < 32; ++kb) {
        // ---- batch-load all 12 B fragments (own registers each) ----
        v16bf bf[12];
#pragma unroll
        for (int mat = 0; mat < 3; ++mat) {
#pragma unroll
            for (int ct = 0; ct < 4; ++ct) {
                const __bf16* wr = wbase + (size_t)mat * (HD * EMBED)
                                         + (size_t)(ct * 16) * EMBED + kb * 32;
                bf[mat * 4 + ct] = cat16(ldg8(wr), ldg8(wr + 8));
            }
        }

        // ---- A fragment from X (fp32 -> bf16 via packed hw cvt) ----
        const float* xp = xrow + kb * 32;
        int e0 = 16 * half;
        v8f x0 = *(const v8f*)(xp + e0);
        v8f x1 = *(const v8f*)(xp + e0 + 8);
        v16bf a = cat16(__builtin_convertvector(x0, v8bf),
                        __builtin_convertvector(x1, v8bf));

        // ---- 12 back-to-back WMMAs ----
#pragma unroll
        for (int mat = 0; mat < 3; ++mat)
#pragma unroll
            for (int ct = 0; ct < 4; ++ct)
                acc[mat][ct] = wmma_bf16(a, bf[mat * 4 + ct], acc[mat][ct]);
    }

    // ---- store: Q,K row-major ; V transposed ----
    __bf16* qp = Qb + ((size_t)(b * SEQ + s0)) * HD;
    __bf16* kp = Kb + ((size_t)(b * SEQ + s0)) * HD;
#pragma unroll
    for (int ct = 0; ct < 4; ++ct) {
        v8bf q8  = __builtin_convertvector(acc[0][ct], v8bf);
        v8bf k8  = __builtin_convertvector(acc[1][ct], v8bf);
        v8bf vv8 = __builtin_convertvector(acc[2][ct], v8bf);
#pragma unroll
        for (int r = 0; r < 8; ++r) {
            int m = r + 8 * half;
            qp[(size_t)m * HD + ct * 16 + n] = q8[r];
            kp[(size_t)m * HD + ct * 16 + n] = k8[r];
            Vt[((size_t)b * HD + ct * 16 + n) * SEQ + s0 + m] = vv8[r];
        }
    }
}

// ---------------------------------------------------------------------------
// Kernel 2: causal flash attention, double-buffered async K/V staging.
// Block = 4 waves = 4 consecutive 16-query tiles of one batch.
// ---------------------------------------------------------------------------
__global__ __launch_bounds__(128)
void attn_kernel(const __bf16* __restrict__ Qb,
                 const __bf16* __restrict__ Kb,
                 const __bf16* __restrict__ Vt,
                 float* __restrict__ out) {
    __shared__ __bf16 sK[2][32][64];    // 2 x 4 KB
    __shared__ __bf16 sV[2][64][32];    // 2 x 4 KB
    __shared__ __bf16 sP[4][16][32];    // per-wave P transpose slab

    int t    = threadIdx.x;
    int lane = t & 31;
    int wid  = t >> 5;
    int n    = lane & 15;
    int half = lane >> 4;

    int tile = blockIdx.x * 4 + wid;      // 0..1023 (all 4 share one batch)
    int b  = tile >> 7;
    int qt = tile & 127;
    int qt_max = (blockIdx.x * 4 + 3) & 127;

    // ---- Q A-fragments (row m = n) ----
    const __bf16* qrow = Qb + ((size_t)(b * SEQ + qt * 16 + n)) * HD;
    v16bf qf[2];
#pragma unroll
    for (int f = 0; f < 2; ++f) {
        int d0 = 32 * f + 16 * half;
        qf[f] = cat16(ldg8(qrow + d0), ldg8(qrow + d0 + 8));
    }

    v8f o[4] = {};
    float mrow[8], lrow[8];
#pragma unroll
    for (int r = 0; r < 8; ++r) { mrow[r] = -3.0e38f; lrow[r] = 0.0f; }

    // all waves run the block-max trip count so barriers stay uniform;
    // surplus key tiles are fully masked and contribute nothing.
    int nkt = (qt_max * 16 + 15) / 32 + 1;

    const __bf16* Kbat = Kb + (size_t)b * SEQ * HD;
    const __bf16* Vbat = Vt + (size_t)b * HD * SEQ;

    // per-thread chunk decomposition for cooperative staging
    auto stage = [&](int kt, int buf) {
        const __bf16* Kbase = Kbat + (size_t)kt * 32 * HD;
        const __bf16* Vbase = Vbat + (size_t)kt * 32;
#pragma unroll
        for (int c0 = 0; c0 < 2; ++c0) {
            int c = t + c0 * 128;                       // 16B chunk id, 0..255
            async_copy16(Kbase + (c >> 3) * HD + (c & 7) * 8,
                         &sK[buf][0][0] + c * 8);
            async_copy16(Vbase + (size_t)(c >> 2) * SEQ + (c & 3) * 8,
                         &sV[buf][0][0] + c * 8);
        }
    };

    stage(0, 0);                          // prologue: tile 0 -> buffer 0

    for (int kt = 0; kt < nkt; ++kt) {
        int cur = kt & 1;
        bool more = (kt + 1 < nkt);
        if (more) stage(kt + 1, cur ^ 1); // overlap next tile's DMA w/ compute

        // async loads complete in order: <=4 outstanding means the current
        // tile's 4 copies (this wave's) have landed.
        if (more) wait_async4(); else wait_async0();
        __syncthreads();

        // ---- S = Q K^T : two 16x16 tiles (keys g*16..) ----
        v8f s[2];
#pragma unroll
        for (int g = 0; g < 2; ++g) {
            const __bf16* krow = &sK[cur][g * 16 + n][0];
            v8f c = {};
#pragma unroll
            for (int h = 0; h < 2; ++h) {
                int d0 = 32 * h + 16 * half;
                v16bf kf = cat16(*(const v8bf*)(krow + d0),
                                 *(const v8bf*)(krow + d0 + 8));
                c = wmma_bf16(qf[h], kf, c);
            }
            s[g] = c;
        }

        // ---- scale + causal mask + online softmax ----
#pragma unroll
        for (int r = 0; r < 8; ++r) {
            int i  = qt * 16 + r + 8 * half;
            int j0 = kt * 32 + n;
            int j1 = kt * 32 + 16 + n;
            float v0 = (j0 <= i) ? s[0][r] * 0.125f : -3.0e38f;
            float v1 = (j1 <= i) ? s[1][r] * 0.125f : -3.0e38f;

            float mx = fmaxf(v0, v1);
            mx = fmaxf(mx, __shfl_xor(mx, 1));
            mx = fmaxf(mx, __shfl_xor(mx, 2));
            mx = fmaxf(mx, __shfl_xor(mx, 4));
            mx = fmaxf(mx, __shfl_xor(mx, 8));

            float mnew  = fmaxf(mrow[r], mx);
            float alpha = __expf(mrow[r] - mnew);
            float p0 = __expf(v0 - mnew);
            float p1 = __expf(v1 - mnew);
            float rs = p0 + p1;
            rs += __shfl_xor(rs, 1);
            rs += __shfl_xor(rs, 2);
            rs += __shfl_xor(rs, 4);
            rs += __shfl_xor(rs, 8);

            lrow[r] = lrow[r] * alpha + rs;
            mrow[r] = mnew;
#pragma unroll
            for (int ct = 0; ct < 4; ++ct) o[ct][r] *= alpha;

            sP[wid][r + 8 * half][n]      = (__bf16)p0;
            sP[wid][r + 8 * half][16 + n] = (__bf16)p1;
        }

        // intra-wave LDS RAW: cross-lane P stores must land before transpose load
        asm volatile("s_wait_dscnt 0" ::: "memory");

        // ---- P A-fragment (row m = n, k = local key 0..31) ----
        v16bf pf;
        {
            const __bf16* pr = &sP[wid][n][0];
            int e0 = 16 * half;
            pf = cat16(*(const v8bf*)(pr + e0), *(const v8bf*)(pr + e0 + 8));
        }

        // ---- O += P x V (sV rows are d-major, keys contiguous) ----
#pragma unroll
        for (int ct = 0; ct < 4; ++ct) {
            const __bf16* vr = &sV[cur][ct * 16 + n][0] + 16 * half;
            v16bf vf = cat16(*(const v8bf*)(vr), *(const v8bf*)(vr + 8));
            o[ct] = wmma_bf16(pf, vf, o[ct]);
        }

        __syncthreads();   // all waves done with buffer `cur` before it is
                           // re-staged at iteration kt+1
    }

    // ---- epilogue: normalize and store fp32 ----
    float* orow = out + ((size_t)(b * SEQ + qt * 16)) * HD;
#pragma unroll
    for (int ct = 0; ct < 4; ++ct) {
#pragma unroll
        for (int r = 0; r < 8; ++r) {
            orow[(size_t)(r + 8 * half) * HD + ct * 16 + n] =
                o[ct][r] / lrow[r];
        }
    }
}

// ---------------------------------------------------------------------------
extern "C" void kernel_launch(void* const* d_in, const int* in_sizes, int n_in,
                              void* d_out, int out_size, void* d_ws, size_t ws_size,
                              hipStream_t stream) {
    const float* X  = (const float*)d_in[0];
    const float* Wq = (const float*)d_in[1];
    const float* Wk = (const float*)d_in[2];
    const float* Wv = (const float*)d_in[3];
    float* out = (float*)d_out;

    char* ws = (char*)d_ws;
    const size_t WT_BYTES  = (size_t)3 * HD * EMBED * sizeof(__bf16);  // 384 KB
    const size_t MAT_BYTES = (size_t)NB * SEQ * HD * sizeof(__bf16);   // 2 MB
    __bf16* WT = (__bf16*)ws;
    __bf16* Qb = (__bf16*)(ws + WT_BYTES);
    __bf16* Kb = (__bf16*)(ws + WT_BYTES + MAT_BYTES);
    __bf16* Vt = (__bf16*)(ws + WT_BYTES + 2 * MAT_BYTES);

    // 3 * 64 * 1024 = 196608 elements
    wtrans_kernel<<<768, 256, 0, stream>>>(Wq, Wk, Wv, WT);

    // B*S/16 = 1024 tiles, 4 waves / block
    qkv_project_kernel<<<256, 128, 0, stream>>>(X, WT, Qb, Kb, Vt);

    attn_kernel<<<256, 128, 0, stream>>>(Qb, Kb, Vt, out);
}
